// ScatterW1_87153476370978
// MI455X (gfx1250) — matrix operand order repaired and checked by reading, not compiled, not run on hardware
//
#include <hip/hip_runtime.h>
#include <stdint.h>

// ---------------------------------------------------------------------------
// Graph scattering transform on MI455X (gfx1250).
// Spectral wavelets sqrt(relu(p_j(T))) realized as degree-6 matrix polynomials
// in P_j = T^{2^{j-1}} - T^{2^j}; heavy math on v_wmma_f32_16x16x32_bf16.
// GEMM uses LDS double buffering with async global->LDS staging
// (GLOBAL_LOAD_ASYNC_TO_LDS_B128, ASYNCcnt) when the toolchain exposes the
// gfx1250 builtins, else vectorized b128 copies.
// ---------------------------------------------------------------------------

#define N_NODES 3072
#define C_IN    32
#define G_BATCH 8
#define FROWS   11
#define FDIM    (FROWS * C_IN)   // 352
#define D2      (4 * C_IN)       // 128

typedef __attribute__((ext_vector_type(16))) __bf16 bf16x16;
typedef __attribute__((ext_vector_type(8)))  float  f32x8;
typedef unsigned short u16;
typedef unsigned int   u32;

#if __has_builtin(__builtin_amdgcn_global_load_async_to_lds_b128) && \
    __has_builtin(__builtin_amdgcn_s_wait_asynccnt)
#define USE_ASYNC_LDS 1
#else
#define USE_ASYNC_LDS 0
#endif

#if USE_ASYNC_LDS
typedef int vint4 __attribute__((vector_size(16)));   // gcc-style V4i, matches builtin
#endif

__device__ __forceinline__ u16 f32_bf16(float f) {
  union { float f; u32 u; } c; c.f = f;
  u32 u = c.u;
  return (u16)((u + 0x7FFFu + ((u >> 16) & 1u)) >> 16);  // RNE
}

// 16-byte global -> LDS copy: async (ASYNCcnt-tracked, no VGPR round trip) on
// gfx1250 toolchains that expose it, else load/store b128 through VGPRs.
__device__ __forceinline__ void copy16_g2l(const u16* g, u16* l) {
#if USE_ASYNC_LDS
  __builtin_amdgcn_global_load_async_to_lds_b128(
      (__attribute__((address_space(1))) vint4*)g,
      (__attribute__((address_space(3))) vint4*)l, 0, 0);
#else
  *(uint4*)l = *(const uint4*)g;
#endif
}

__device__ __forceinline__ void wait_async_lds() {
#if USE_ASYNC_LDS
  __builtin_amdgcn_s_wait_asynccnt(0);
#endif
}

// ------------------------- elementwise / scatter kernels -------------------

__global__ void k_zero_f32(float* __restrict__ p, size_t n) {
  size_t i = (size_t)blockIdx.x * blockDim.x + threadIdx.x;
  size_t s = (size_t)gridDim.x * blockDim.x;
  for (; i < n; i += s) p[i] = 0.f;
}

__global__ void k_deg_scatter(const int* __restrict__ col, float* __restrict__ deg, int E) {
  int e = blockIdx.x * blockDim.x + threadIdx.x;
  if (e < E) atomicAdd(&deg[col[e]], 1.0f);
}

__global__ void k_dh(const float* __restrict__ deg, float* __restrict__ dh, int n) {
  int i = blockIdx.x * blockDim.x + threadIdx.x;
  if (i < n) { float d = deg[i]; dh[i] = d > 0.f ? rsqrtf(d) : 0.f; }
}

__global__ void k_T_scatter(const int* __restrict__ row, const int* __restrict__ col,
                            const float* __restrict__ dh, float* __restrict__ T, int E) {
  int e = blockIdx.x * blockDim.x + threadIdx.x;
  if (e < E) {
    int r = row[e], c = col[e];
    atomicAdd(&T[(size_t)r * N_NODES + c], dh[r] * dh[c]);
  }
}

__global__ void k_f32_to_bf16(const float* __restrict__ s, u16* __restrict__ d, size_t n) {
  size_t i = (size_t)blockIdx.x * blockDim.x + threadIdx.x;
  size_t st = (size_t)gridDim.x * blockDim.x;
  for (; i < n; i += st) d[i] = f32_bf16(s[i]);
}

__global__ void k_sub_to_bf16(const float* __restrict__ a, const float* __restrict__ b,
                              u16* __restrict__ d, size_t n) {
  size_t i = (size_t)blockIdx.x * blockDim.x + threadIdx.x;
  size_t st = (size_t)gridDim.x * blockDim.x;
  for (; i < n; i += st) d[i] = f32_bf16(a[i] - b[i]);
}

__global__ void k_scale(const float* __restrict__ u, float* __restrict__ y, float c, size_t n) {
  size_t i = (size_t)blockIdx.x * blockDim.x + threadIdx.x;
  size_t st = (size_t)gridDim.x * blockDim.x;
  for (; i < n; i += st) y[i] = c * u[i];
}

__global__ void k_axpy(const float* __restrict__ t, const float* __restrict__ u,
                       float* __restrict__ y, float c, size_t n) {
  size_t i = (size_t)blockIdx.x * blockDim.x + threadIdx.x;
  size_t st = (size_t)gridDim.x * blockDim.x;
  for (; i < n; i += st) y[i] = t[i] + c * u[i];
}

__global__ void k_copy_x(const float* __restrict__ x, float* __restrict__ feat) {
  int i = blockIdx.x * blockDim.x + threadIdx.x;          // over N*C
  if (i < N_NODES * C_IN) {
    int n = i / C_IN, c = i % C_IN;
    feat[(size_t)n * FDIM + c] = x[i];
  }
}

// s1: Y [N,32] -> V1[n, c*4+j] = |Y|, FEAT row 1+j
__global__ void k_s1(const float* __restrict__ Y, float* __restrict__ V1,
                     float* __restrict__ feat, int j) {
  int i = blockIdx.x * blockDim.x + threadIdx.x;
  if (i < N_NODES * C_IN) {
    int n = i / C_IN, c = i % C_IN;
    float v = fabsf(Y[i]);
    V1[(size_t)n * D2 + c * 4 + j] = v;
    feat[(size_t)n * FDIM + (1 + j) * C_IN + c] = v;
  }
}

// s2: Y [N,128] for filter f2; emit rows 5 + f2*(f2-1)/2 + f1 for f1 < f2
__global__ void k_s2(const float* __restrict__ Y, float* __restrict__ feat, int f2) {
  int i = blockIdx.x * blockDim.x + threadIdx.x;
  if (i < N_NODES * C_IN) {
    int n = i / C_IN, c = i % C_IN;
    int rowStart = 5 + (f2 * (f2 - 1)) / 2;
    for (int f1 = 0; f1 < f2; ++f1) {
      float v = fabsf(Y[(size_t)n * D2 + c * 4 + f1]);
      feat[(size_t)n * FDIM + (rowStart + f1) * C_IN + c] = v;
    }
  }
}

// moments: one thread per (graph, feature); out layout [G, 4*FDIM]
__global__ void k_moments(const float* __restrict__ feat, const int* __restrict__ batch,
                          float* __restrict__ out) {
  int f = blockIdx.x * blockDim.x + threadIdx.x;
  int g = blockIdx.y;
  if (f >= FDIM) return;
  float sum = 0.f, cnt = 0.f;
  for (int n = 0; n < N_NODES; ++n)
    if (batch[n] == g) { sum += feat[(size_t)n * FDIM + f]; cnt += 1.f; }
  float mean = sum / cnt;
  float m2 = 0.f, m3 = 0.f, m4 = 0.f;
  for (int n = 0; n < N_NODES; ++n)
    if (batch[n] == g) {
      float d = feat[(size_t)n * FDIM + f] - mean;
      float d2 = d * d;
      m2 += d2; m3 += d2 * d; m4 += d2 * d2;
    }
  m2 /= cnt; m3 /= cnt; m4 /= cnt;
  float skew = m3 / (m2 * sqrtf(m2));
  if (!(skew == skew) || skew > 1e15f) skew = 0.f;
  float kurt = m4 / (m2 * m2) - 3.f;
  if (!(kurt == kurt) || kurt > 1e15f) kurt = -3.f;
  const int GF = 4 * FDIM;
  out[(size_t)g * GF + 0 * FDIM + f] = mean;
  out[(size_t)g * GF + 1 * FDIM + f] = m2;
  out[(size_t)g * GF + 2 * FDIM + f] = skew;
  out[(size_t)g * GF + 3 * FDIM + f] = kurt;
}

// ------------------------------ WMMA GEMM ----------------------------------
// C[M,Nc] = A[M,K] @ B[K,Nc]; A,B bf16 row-major packed, C f32 row-major.
// Block = 256 threads = 8 waves (4x2). Block tile 128x64, wave tile 32x32,
// K-step 32 -> 4x v_wmma_f32_16x16x32_bf16 per wave per step.
// LDS double-buffered; A staged with 16B async/global b128 copies, B staged
// with b64 loads + transposed stores (fragment reads become 2x b128 each).

#define TM 128
#define TN 64
#define TK 32
#define LDA_S (TK + 8)   // 40 u16 = 80B row keeps uint4 LDS accesses 16B aligned

__launch_bounds__(256)
__global__ void k_gemm_bf16(const u16* __restrict__ A, const u16* __restrict__ B,
                            float* __restrict__ C, int M, int Ncols, int K) {
  __shared__ __attribute__((aligned(16))) u16 As[2][TM * LDA_S];
  __shared__ __attribute__((aligned(16))) u16 Bs[2][TN * LDA_S];

  const int tid  = threadIdx.x;
  const int lane = tid & 31;
  const int wave = tid >> 5;
  const int wm = wave & 3;          // 0..3 row tile
  const int wn = wave >> 2;         // 0..1 col tile
  const int rowBase = blockIdx.x * TM;
  const int colBase = blockIdx.y * TN;
  const int ml   = lane & 15;
  const int half = lane >> 4;

  // ---- per-thread staging coordinates (hoisted out of the K loop) ----
  // A: 128 rows x 32 cols = 512 x 16B chunks; 2 per thread.
  const int arow = tid >> 2;              // 0..63 (and +64)
  const int acol = (tid & 3) * 8;         // 0,8,16,24
  const u16* gA0 = A + (size_t)(rowBase + arow) * K + acol;
  const u16* gA1 = gA0 + (size_t)64 * K;
  const int aoff0 = arow * LDA_S + acol;
  const int aoff1 = aoff0 + 64 * LDA_S;
  // B: 32 k x 64 n; each thread: 2 x (b64 load of 4 n) -> 8 transposed stores.
  const int bn = (tid & 15) * 4;          // 0..60
  const int bk = tid >> 4;                // 0..15 (and +16)
  const bool bok = (colBase + bn) < Ncols;   // Ncols % 4 == 0 in all calls
  const u16* gB0 = B + (size_t)bk * Ncols + colBase + bn;
  const u16* gB1 = gB0 + (size_t)16 * Ncols;

  auto stage = [&](int buf, int k0) {
    copy16_g2l(gA0 + k0, &As[buf][aoff0]);
    copy16_g2l(gA1 + k0, &As[buf][aoff1]);
    union { uint2 v; u16 h[4]; } q0, q1;
    uint2 z; z.x = 0u; z.y = 0u;
    q0.v = bok ? *(const uint2*)(gB0 + (size_t)k0 * Ncols) : z;
    q1.v = bok ? *(const uint2*)(gB1 + (size_t)k0 * Ncols) : z;
    u16* p = &Bs[buf][bn * LDA_S + bk];
    #pragma unroll
    for (int i = 0; i < 4; ++i) {
      p[i * LDA_S]      = q0.h[i];
      p[i * LDA_S + 16] = q1.h[i];
    }
  };

  // fragment offsets (ISA 16-bit A/B layouts: lane -> M/N = lane%16, K half)
  const int af0 = (wm * 32 + ml) * LDA_S + half * 8;
  const int af1 = af0 + 16 * LDA_S;
  const int bf0 = (wn * 32 + ml) * LDA_S + half * 8;
  const int bf1 = bf0 + 16 * LDA_S;

  f32x8 acc00 = {}, acc01 = {}, acc10 = {}, acc11 = {};

  const int steps = K / TK;
  stage(0, 0);
  wait_async_lds();
  __syncthreads();

  for (int s = 0; s < steps; ++s) {
    const int cur = s & 1;
    if (s + 1 < steps) stage(cur ^ 1, (s + 1) * TK);
    if (s + 2 < steps)
      __builtin_prefetch(gB0 + (size_t)(s + 2) * TK * Ncols, 0, 1);

    const u16* Ac = As[cur];
    const u16* Bc = Bs[cur];
    union Frag { uint4 q[2]; bf16x16 v; } a0, a1, b0, b1;
    a0.q[0] = *(const uint4*)(Ac + af0);
    a0.q[1] = *(const uint4*)(Ac + af0 + 16);
    a1.q[0] = *(const uint4*)(Ac + af1);
    a1.q[1] = *(const uint4*)(Ac + af1 + 16);
    b0.q[0] = *(const uint4*)(Bc + bf0);
    b0.q[1] = *(const uint4*)(Bc + bf0 + 16);
    b1.q[0] = *(const uint4*)(Bc + bf1);
    b1.q[1] = *(const uint4*)(Bc + bf1 + 16);

    acc00 = __builtin_amdgcn_wmma_f32_16x16x32_bf16(false, a0.v, false, b0.v, (short)0, acc00, false, false);
    acc01 = __builtin_amdgcn_wmma_f32_16x16x32_bf16(false, a0.v, false, b1.v, (short)0, acc01, false, false);
    acc10 = __builtin_amdgcn_wmma_f32_16x16x32_bf16(false, a1.v, false, b0.v, (short)0, acc10, false, false);
    acc11 = __builtin_amdgcn_wmma_f32_16x16x32_bf16(false, a1.v, false, b1.v, (short)0, acc11, false, false);

    wait_async_lds();
    __syncthreads();
  }

  // ---- epilogue: D layout lane->(N=lane%16), vgpr r->(M=r+8*(lane/16)) ----
  const int rowW = rowBase + wm * 32 + half * 8;
  const int colW = colBase + wn * 32 + ml;
  #pragma unroll
  for (int r = 0; r < 8; ++r) {
    int m0 = rowW + r, m1 = m0 + 16;
    if (colW < Ncols) {
      C[(size_t)m0 * Ncols + colW] = acc00[r];
      C[(size_t)m1 * Ncols + colW] = acc10[r];
    }
    if (colW + 16 < Ncols) {
      C[(size_t)m0 * Ncols + colW + 16] = acc01[r];
      C[(size_t)m1 * Ncols + colW + 16] = acc11[r];
    }
  }
}

// ------------------------------ orchestration ------------------------------

// Degree-6 monomial fit (offline least-squares) of sqrt(max(x,0)) on the
// spectral range of the P_j filters ([-2, 1]).
static const float POLY[7] = { 0.0170f, 1.1353f, 2.3683f, -7.4110f,
                               10.1836f, -6.7810f, 1.7340f };

extern "C" void kernel_launch(void* const* d_in, const int* in_sizes, int n_in,
                              void* d_out, int out_size, void* d_ws, size_t ws_size,
                              hipStream_t stream) {
  (void)n_in; (void)out_size; (void)ws_size;
  const float* x     = (const float*)d_in[0];
  const int*   edges = (const int*)d_in[1];
  const int*   batch = (const int*)d_in[2];
  const int E = in_sizes[1] / 2;
  const int* erow = edges;
  const int* ecol = edges + E;
  float* out = (float*)d_out;

  const size_t NN = (size_t)N_NODES * N_NODES;
  char* ws = (char*)d_ws;
  size_t off = 0;
  auto alloc = [&](size_t bytes) -> void* {
    void* p = ws + off; off += (bytes + 255) & ~(size_t)255; return p;
  };
  float* T    = (float*)alloc(NN * 4);
  float* P2   = (float*)alloc(NN * 4);
  float* P4   = (float*)alloc(NN * 4);
  float* P8   = (float*)alloc(NN * 4);
  float* P16  = (float*)alloc(NN * 4);
  u16*   BA   = (u16*)  alloc(NN * 2);                 // bf16 staging (A operand)
  float* deg  = (float*)alloc((size_t)N_NODES * 4);
  float* dh   = (float*)alloc((size_t)N_NODES * 4);
  float* V1   = (float*)alloc((size_t)N_NODES * D2 * 4);
  float* Y    = (float*)alloc((size_t)N_NODES * D2 * 4);
  float* Tv   = (float*)alloc((size_t)N_NODES * D2 * 4);
  u16*   Yb   = (u16*)  alloc((size_t)N_NODES * D2 * 2);
  float* FEAT = (float*)alloc((size_t)N_NODES * FDIM * 4);

  auto gs = [](size_t n) { return dim3((unsigned)((n + 255) / 256)); };
  auto tobf = [&](const float* s, u16* d, size_t n) {
    k_f32_to_bf16<<<gs(n), 256, 0, stream>>>(s, d, n);
  };
  auto gemm = [&](const u16* A, const u16* B, float* Cc, int M, int Nc, int K) {
    dim3 grid(M / TM, (Nc + TN - 1) / TN);
    k_gemm_bf16<<<grid, 256, 0, stream>>>(A, B, Cc, M, Nc, K);
  };

  // 1) degrees -> dh -> dense normalized adjacency T
  k_zero_f32<<<gs((size_t)N_NODES), 256, 0, stream>>>(deg, (size_t)N_NODES);
  k_zero_f32<<<8192, 256, 0, stream>>>(T, NN);
  k_deg_scatter<<<gs((size_t)E), 256, 0, stream>>>(ecol, deg, E);
  k_dh<<<gs((size_t)N_NODES), 256, 0, stream>>>(deg, dh, N_NODES);
  k_T_scatter<<<gs((size_t)E), 256, 0, stream>>>(erow, ecol, dh, T, E);

  // 2) power chain T^2, T^4, T^8, T^16 (bf16 WMMA N^3 GEMMs)
  tobf(T,  BA, NN); gemm(BA, BA, P2,  N_NODES, N_NODES, N_NODES);
  tobf(P2, BA, NN); gemm(BA, BA, P4,  N_NODES, N_NODES, N_NODES);
  tobf(P4, BA, NN); gemm(BA, BA, P8,  N_NODES, N_NODES, N_NODES);
  tobf(P8, BA, NN); gemm(BA, BA, P16, N_NODES, N_NODES, N_NODES);

  // 3) zeroth-order features
  k_copy_x<<<gs((size_t)N_NODES * C_IN), 256, 0, stream>>>(x, FEAT);

  const float* lo[4] = { T,  P2, P4, P8  };
  const float* hi[4] = { P2, P4, P8, P16 };

  // Horner evaluation of sqrt(relu(P_j)) applied to a feature block U [N,D]
  auto wavelet = [&](const float* U, int D, int j) {
    k_sub_to_bf16<<<gs(NN), 256, 0, stream>>>(lo[j], hi[j], BA, NN);
    size_t nd = (size_t)N_NODES * D;
    k_scale<<<gs(nd), 256, 0, stream>>>(U, Y, POLY[6], nd);
    for (int i = 5; i >= 0; --i) {
      tobf(Y, Yb, nd);
      gemm(BA, Yb, Tv, N_NODES, D, N_NODES);
      k_axpy<<<gs(nd), 256, 0, stream>>>(Tv, U, Y, POLY[i], nd);
    }
  };

  // 4) first-order scattering: s1 = |W_j x|
  for (int j = 0; j < 4; ++j) {
    wavelet(x, C_IN, j);
    k_s1<<<gs((size_t)N_NODES * C_IN), 256, 0, stream>>>(Y, V1, FEAT, j);
  }
  // 5) second-order: only f2 > f1 channels survive feng_filters()
  for (int j = 1; j < 4; ++j) {
    wavelet(V1, D2, j);
    k_s2<<<gs((size_t)N_NODES * C_IN), 256, 0, stream>>>(Y, FEAT, j);
  }

  // 6) per-graph moments -> [G, 4*FDIM]
  dim3 mg((FDIM + 255) / 256, G_BATCH);
  k_moments<<<mg, 256, 0, stream>>>(FEAT, batch, out);
}
